// MultiQueryAttention_996432413522
// MI455X (gfx1250) — matrix operand order repaired and checked by reading, not compiled, not run on hardware
//
#include <hip/hip_runtime.h>
#include <hip/hip_bf16.h>
#include <math.h>

// ---------------- shapes ----------------
#define HIDDEN 2048
#define NH     16
#define NKV    4
#define HD     128
#define BATCH  2
#define SEQ    2048
#define MROWS  (BATCH * SEQ)   // 4096

typedef __attribute__((ext_vector_type(16))) __bf16 bf16x16;
typedef __attribute__((ext_vector_type(8)))  float  f32x8;
typedef __attribute__((ext_vector_type(4)))  unsigned int u32x4;
typedef __attribute__((ext_vector_type(8)))  unsigned int u32x8;

// fp32 -> bf16 (round to nearest even)
static __device__ __forceinline__ unsigned short f2bf(float f) {
  unsigned int u = __float_as_uint(f);
  u += 0x7FFFu + ((u >> 16) & 1u);
  return (unsigned short)(u >> 16);
}

// D(16x16,f32) += A(16x32,bf16) * B(32x16,bf16)
static __device__ __forceinline__ void wmma_bf16_acc(const unsigned int a[8],
                                                     const unsigned int b[8],
                                                     float c[8]) {
  bf16x16 av, bv; f32x8 cv;
  __builtin_memcpy(&av, a, 32);
  __builtin_memcpy(&bv, b, 32);
  __builtin_memcpy(&cv, c, 32);
  cv = __builtin_amdgcn_wmma_f32_16x16x32_bf16(false, av, false, bv,
                                               (short)0, cv, false, false);
  __builtin_memcpy(c, &cv, 32);
}

// K-offset inside a 16x32 bf16 A fragment for dword j, lane-half lh (ISA 7.12.2)
static __device__ __forceinline__ int a_koff(int j, int lh) {
  return (j < 4 ? 2 * j : 16 + 2 * (j - 4)) + 8 * lh;
}

// LDS byte offset of a __shared__ address (low 32 bits of the flat pointer)
static __device__ __forceinline__ unsigned lds_off(const void* p) {
  return (unsigned)(uintptr_t)p;
}

// CDNA5 async global->LDS copy, 16 bytes per lane, tracked by ASYNCcnt.
static __device__ __forceinline__ void async_ld_b128(unsigned lds_byte_off,
                                                     const void* gptr) {
  asm volatile("global_load_async_to_lds_b128 %0, %1, off"
               :: "v"(lds_byte_off),
                  "v"((unsigned long long)(uintptr_t)gptr)
               : "memory");
}
#define WAIT_ASYNC(n) asm volatile("s_wait_asynccnt " #n ::: "memory")

// CDNA5 Tensor Data Mover: one 2D tile Global->LDS DMA, tracked by TENSORcnt.
// D# group0/group1 packed per CDNA5 ISA (2D: groups 2/3 NULL).
// g1flags carries data_size + pad_enable + pad_interval + pad_amount.
static __device__ __forceinline__ void tdm_load_2d(unsigned lds_byte_off,
                                                   unsigned long long gaddr,
                                                   unsigned g1flags,
                                                   unsigned tensor_d0,
                                                   unsigned tensor_d1,
                                                   unsigned tile_d0,
                                                   unsigned tile_d1,
                                                   unsigned stride0) {
  u32x4 g0;
  g0.x = 1u;                                   // count=1, user descriptor
  g0.y = lds_byte_off;                         // lds_addr [63:32]
  g0.z = (unsigned)gaddr;                      // global_addr [95:64]
  g0.w = ((unsigned)(gaddr >> 32) & 0x01FFFFFFu) | (2u << 30);  // addr[56:32] | type=2
  u32x8 g1;
  g1.s0 = g1flags;                             // mask=0 | data_size | pad cfg
  g1.s1 = (tensor_d0 & 0xFFFFu) << 16;         // tensor_dim0[15:0] -> bits 63:48
  g1.s2 = (tensor_d0 >> 16) | ((tensor_d1 & 0xFFFFu) << 16);   // td0 hi | td1 lo
  g1.s3 = (tensor_d1 >> 16) | (tile_d0 << 16); // td1 hi | tile_dim0
  g1.s4 = tile_d1;                             // tile_dim1 | tile_dim2=0
  g1.s5 = stride0;                             // tensor_dim0_stride[31:0]
  g1.s6 = 0u;                                  // stride0 hi | stride1 lo
  g1.s7 = 0u;                                  // stride1 hi
  asm volatile("tensor_load_to_lds %0, %1" :: "s"(g0), "s"(g1) : "memory");
}
// data_size=2B (1<<16), pad_enable (1<<20), pad_interval n (2^(n+1) dwords), pad 4 dwords (3<<25)
#define TDM_FLAGS_256B_ROW ((1u << 16) | (1u << 20) | (5u << 22) | (3u << 25))
#define TDM_FLAGS_128B_ROW ((1u << 16) | (1u << 20) | (4u << 22) | (3u << 25))

// ---------------- elementwise fp32 -> bf16 ----------------
__global__ void cvt_bf16(const float* __restrict__ in,
                         unsigned short* __restrict__ out, size_t n) {
  size_t i = (size_t)blockIdx.x * blockDim.x + threadIdx.x;
  size_t stride = (size_t)gridDim.x * blockDim.x;
  for (; i < n; i += stride) out[i] = f2bf(in[i]);
}

// ---------------- fp32 [K,N] -> bf16 transposed [N,K] ----------------
__global__ void cvt_bf16_t(const float* __restrict__ in,
                           unsigned short* __restrict__ out, int K, int N) {
  size_t total = (size_t)K * N;
  size_t i = (size_t)blockIdx.x * blockDim.x + threadIdx.x;
  size_t stride = (size_t)gridDim.x * blockDim.x;
  for (; i < total; i += stride) {
    int k = (int)(i / N);
    int n = (int)(i % N);
    out[(size_t)n * K + k] = f2bf(in[i]);
  }
}

// -------- RoPE + bf16: [B,S,H,D] -> [B,H,S,D]  (or d-major [B,H,D,S]) -------
__global__ void rope_cvt(const float* __restrict__ in,
                         unsigned short* __restrict__ out,
                         int nheads, float scale, int doRope, int dmajor,
                         int total) {
  int idx = blockIdx.x * blockDim.x + threadIdx.x;
  if (idx >= total) return;
  int i    = idx & 63;
  int h    = (idx >> 6) % nheads;
  int rest = (idx >> 6) / nheads;
  int s    = rest % SEQ;
  int bb   = rest / SEQ;
  const float* row = in + (((size_t)(bb * SEQ + s)) * nheads + h) * HD;
  float x1 = row[i], x2 = row[i + 64];
  float o1 = x1, o2 = x2;
  if (doRope) {
    float invf = __powf(10000.0f, -(float)i * (1.0f / 64.0f));
    float ang  = (float)s * invf;
    float cs = __cosf(ang), sn = __sinf(ang);
    o1 = x1 * cs - x2 * sn;
    o2 = x2 * cs + x1 * sn;
  }
  if (dmajor) {
    size_t base = ((size_t)(bb * nheads + h)) * HD;
    out[(base + i) * SEQ + s]      = f2bf(o1 * scale);
    out[(base + i + 64) * SEQ + s] = f2bf(o2 * scale);
  } else {
    size_t ob = (((size_t)(bb * nheads + h)) * SEQ + s) * HD + i;
    out[ob]      = f2bf(o1 * scale);
    out[ob + 64] = f2bf(o2 * scale);
  }
}

// ------------- TN bf16 GEMM: C[M,N] = A[M,K] x Bt[N,K]^T -------------------
// block tile 64x128, BK=64, 256 threads = 8 waves (2x4), 32x32 per wave,
// double-buffered LDS filled with GLOBAL_LOAD_ASYNC_TO_LDS_B128.
#define ASTR 72   // padded row stride (bf16 units), 144B = 9 x 16B chunks
#define BSTR 72
__global__ __launch_bounds__(256) void gemm_bf16_tn(const unsigned short* __restrict__ A,
                                                    const unsigned short* __restrict__ Bt,
                                                    float* __restrict__ C,
                                                    int M, int N, int K) {
  __shared__ __align__(16) unsigned short Alds[2][64 * ASTR];
  __shared__ __align__(16) unsigned short Blds[2][128 * BSTR];
  const int tid  = threadIdx.x;
  const int lane = tid & 31, wvid = tid >> 5;
  const int wm = wvid & 1, wn = wvid >> 1;
  const int lm = lane & 15, lh = lane >> 4;
  const int m0 = blockIdx.x * 64, n0 = blockIdx.y * 128;

  const char* Ab  = (const char*)A;
  const char* Btb = (const char*)Bt;

  float acc[2][2][8];
#pragma unroll
  for (int a = 0; a < 2; ++a)
#pragma unroll
    for (int bj = 0; bj < 2; ++bj)
#pragma unroll
      for (int j = 0; j < 8; ++j) acc[a][bj][j] = 0.f;

  // stage one 64(K)-deep tile pair into buffer `buf`: 6 async ops per thread
  auto stage = [&](int buf, int k0) {
#pragma unroll
    for (int i = 0; i < 2; ++i) {   // A: 64 rows x 128B = 512 chunks
      int u = tid + 256 * i;
      int r = u >> 3, c = u & 7;
      async_ld_b128(lds_off(&Alds[buf][r * ASTR + c * 8]),
                    Ab + ((size_t)(m0 + r) * K + k0) * 2 + c * 16);
    }
#pragma unroll
    for (int i = 0; i < 4; ++i) {   // Bt: 128 rows x 128B = 1024 chunks
      int u = tid + 256 * i;
      int n = u >> 3, c = u & 7;
      async_ld_b128(lds_off(&Blds[buf][n * BSTR + c * 8]),
                    Btb + ((size_t)(n0 + n) * K + k0) * 2 + c * 16);
    }
  };

  stage(0, 0);
  const int T = K >> 6;
  for (int kt = 0; kt < T; ++kt) {
    const int cur = kt & 1;
    if (kt + 1 < T) {
      stage(cur ^ 1, (kt + 1) << 6);   // prefetch next tile
      WAIT_ASYNC(0x6);                 // current tile's 6 ops retired (in-order)
    } else {
      WAIT_ASYNC(0x0);
    }
    __syncthreads();

#pragma unroll
    for (int ks = 0; ks < 2; ++ks) {   // two 32-deep WMMA steps
      unsigned int afr[2][8], bfr[2][8];
#pragma unroll
      for (int s = 0; s < 2; ++s) {
        int m = wm * 32 + s * 16 + lm;
#pragma unroll
        for (int j = 0; j < 8; ++j)
          afr[s][j] = *(const unsigned int*)&Alds[cur][m * ASTR + 32 * ks + a_koff(j, lh)];
        int n = wn * 32 + s * 16 + lm;
#pragma unroll
        for (int j = 0; j < 8; ++j)
          bfr[s][j] = *(const unsigned int*)&Blds[cur][n * BSTR + 32 * ks + a_koff(j, lh)];
      }
#pragma unroll
      for (int i = 0; i < 2; ++i)
#pragma unroll
        for (int jn = 0; jn < 2; ++jn)
          wmma_bf16_acc(afr[i], bfr[jn], acc[i][jn]);
    }
    __syncthreads();
  }

#pragma unroll
  for (int si = 0; si < 2; ++si)
#pragma unroll
    for (int sj = 0; sj < 2; ++sj)
#pragma unroll
      for (int jj = 0; jj < 8; ++jj) {
        int r  = m0 + wm * 32 + si * 16 + jj + 8 * lh;
        int cc = n0 + wn * 32 + sj * 16 + lm;
        C[(size_t)r * N + cc] = acc[si][sj][jj];
      }
}

// ---------------- flash attention (causal, GQA), 64 q-rows per block ----------
// K/V tiles DMA'd by the Tensor Data Mover (one tensor_load_to_lds per tile,
// hardware-padded LDS rows), double-buffered, issued by wave 0 only.
#define KSTR 136   // 272B = 256B row + 16B TDM pad
#define VSTR 72    // 144B = 128B row + 16B TDM pad
#define PSTR 72
__global__ __launch_bounds__(128) void attn_fwd(const unsigned short* __restrict__ Qr,
                                                const unsigned short* __restrict__ Kr,
                                                const unsigned short* __restrict__ Vt,
                                                unsigned short* __restrict__ O) {
  __shared__ __align__(16) unsigned short Klds[2][64 * KSTR];
  __shared__ __align__(16) unsigned short Vlds[2][128 * VSTR];
  __shared__ __align__(16) unsigned short Plds[64 * PSTR];

  const int tid  = threadIdx.x;
  const int lane = tid & 31, wv = tid >> 5;
  const int lm = lane & 15, lh = lane >> 4;
  const int q0 = blockIdx.x * 64;
  const int hq = blockIdx.y, b = blockIdx.z;
  const int hk = hq >> 2;

  const unsigned short* Qbase = Qr + ((size_t)(b * NH  + hq)) * SEQ * HD;
  const unsigned long long Kg = (unsigned long long)(uintptr_t)(Kr + ((size_t)(b * NKV + hk)) * SEQ * HD);
  const unsigned long long Vg = (unsigned long long)(uintptr_t)(Vt + ((size_t)(b * NKV + hk)) * HD * SEQ);

  // Q fragments: 16 rows x 128 d per wave, resident in VGPRs
  unsigned int qf[4][8];
  const int mrow = q0 + wv * 16 + lm;
#pragma unroll
  for (int ds = 0; ds < 4; ++ds)
#pragma unroll
    for (int j = 0; j < 8; ++j)
      qf[ds][j] = *(const unsigned int*)(Qbase + (size_t)mrow * HD + 32 * ds + a_koff(j, lh));

  float oacc[8][8];
#pragma unroll
  for (int dt = 0; dt < 8; ++dt)
#pragma unroll
    for (int j = 0; j < 8; ++j) oacc[dt][j] = 0.f;
  float mst[8], lst[8];
#pragma unroll
  for (int j = 0; j < 8; ++j) { mst[j] = -1e30f; lst[j] = 0.f; }

  // TDM stage of one 64-key tile: K 64x128 rows of 256B, V 128x64 rows of 128B
  auto stage = [&](int buf, int k0) {
    tdm_load_2d(lds_off(&Klds[buf][0]), Kg + (size_t)k0 * HD * 2,
                TDM_FLAGS_256B_ROW,
                /*tensor_d0=*/HD, /*tensor_d1=*/SEQ,
                /*tile_d0=*/HD, /*tile_d1=*/64, /*stride0=*/HD);
    tdm_load_2d(lds_off(&Vlds[buf][0]), Vg + (size_t)k0 * 2,
                TDM_FLAGS_128B_ROW,
                /*tensor_d0=*/SEQ, /*tensor_d1=*/HD,
                /*tile_d0=*/64, /*tile_d1=*/HD, /*stride0=*/SEQ);
  };

  if (wv == 0) stage(0, 0);
  const int T = (q0 >> 6) + 1;
  for (int t = 0; t < T; ++t) {
    const int cur = t & 1;
    const int k0  = t << 6;
    if (wv == 0) {
      if (t + 1 < T) {
        stage(cur ^ 1, (t + 1) << 6);                 // prefetch next tile
        __builtin_amdgcn_s_wait_tensorcnt((short)2);  // current tile done (in-order)
      } else {
        __builtin_amdgcn_s_wait_tensorcnt((short)0);
      }
    }
    __syncthreads();

    // S = Q * K^T  (Q pre-scaled by 1/sqrt(D))
    float sc[4][8];
#pragma unroll
    for (int nt = 0; nt < 4; ++nt) {
      float c[8] = {0, 0, 0, 0, 0, 0, 0, 0};
#pragma unroll
      for (int ds = 0; ds < 4; ++ds) {
        unsigned int bfr[8];
#pragma unroll
        for (int j = 0; j < 8; ++j)
          bfr[j] = *(const unsigned int*)&Klds[cur][(nt * 16 + lm) * KSTR + 32 * ds + a_koff(j, lh)];
        wmma_bf16_acc(qf[ds], bfr, c);
      }
#pragma unroll
      for (int j = 0; j < 8; ++j) sc[nt][j] = c[j];
    }

    if (k0 == q0) {   // causal mask, diagonal tile only
#pragma unroll
      for (int nt = 0; nt < 4; ++nt)
#pragma unroll
        for (int j = 0; j < 8; ++j)
          if (nt * 16 + lm > wv * 16 + j + 8 * lh) sc[nt][j] = -1e30f;
    }

    // online softmax (reductions stay inside 16-lane half-groups)
    float alpha[8];
#pragma unroll
    for (int j = 0; j < 8; ++j) {
      float rm = fmaxf(fmaxf(sc[0][j], sc[1][j]), fmaxf(sc[2][j], sc[3][j]));
      rm = fmaxf(rm, __shfl_xor(rm, 1));
      rm = fmaxf(rm, __shfl_xor(rm, 2));
      rm = fmaxf(rm, __shfl_xor(rm, 4));
      rm = fmaxf(rm, __shfl_xor(rm, 8));
      float mn = fmaxf(mst[j], rm);
      alpha[j] = __expf(mst[j] - mn);
      mst[j] = mn;
      float rs = 0.f;
#pragma unroll
      for (int nt = 0; nt < 4; ++nt) {
        float p = __expf(sc[nt][j] - mn);
        sc[nt][j] = p;
        rs += p;
      }
      rs += __shfl_xor(rs, 1);
      rs += __shfl_xor(rs, 2);
      rs += __shfl_xor(rs, 4);
      rs += __shfl_xor(rs, 8);
      lst[j] = lst[j] * alpha[j] + rs;
    }
#pragma unroll
    for (int dt = 0; dt < 8; ++dt)
#pragma unroll
      for (int j = 0; j < 8; ++j) oacc[dt][j] *= alpha[j];

    // P -> LDS re-swizzle (wave-private rows; DS ops are in-order per wave)
#pragma unroll
    for (int nt = 0; nt < 4; ++nt)
#pragma unroll
      for (int j = 0; j < 8; ++j)
        Plds[(wv * 16 + j + 8 * lh) * PSTR + nt * 16 + lm] = f2bf(sc[nt][j]);

    // O += P(16x64) * V(64x128)
#pragma unroll
    for (int kk = 0; kk < 2; ++kk) {
      unsigned int af[8];
#pragma unroll
      for (int j = 0; j < 8; ++j)
        af[j] = *(const unsigned int*)&Plds[(wv * 16 + lm) * PSTR + kk * 32 + a_koff(j, lh)];
#pragma unroll
      for (int dt = 0; dt < 8; ++dt) {
        unsigned int bfr[8];
#pragma unroll
        for (int j = 0; j < 8; ++j)
          bfr[j] = *(const unsigned int*)&Vlds[cur][(dt * 16 + lm) * VSTR + kk * 32 + a_koff(j, lh)];
        wmma_bf16_acc(af, bfr, oacc[dt]);
      }
    }
    __syncthreads();
  }

  // normalize + store O as [B,S,H,D] bf16
#pragma unroll
  for (int j = 0; j < 8; ++j) {
    float inv = 1.0f / lst[j];
    int row = q0 + wv * 16 + j + 8 * lh;
#pragma unroll
    for (int dt = 0; dt < 8; ++dt)
      O[(((size_t)(b * SEQ + row)) * NH + hq) * HD + dt * 16 + lm] =
          f2bf(oacc[dt][j] * inv);
  }
}

// ---------------- host launcher ----------------
extern "C" void kernel_launch(void* const* d_in, const int* in_sizes, int n_in,
                              void* d_out, int out_size, void* d_ws, size_t ws_size,
                              hipStream_t stream) {
  (void)in_sizes; (void)n_in; (void)out_size; (void)ws_size;
  const float* X  = (const float*)d_in[0];
  const float* Wq = (const float*)d_in[1];
  const float* Wk = (const float*)d_in[2];
  const float* Wv = (const float*)d_in[3];
  const float* Wo = (const float*)d_in[4];
  float* out = (float*)d_out;

  char* ws = (char*)d_ws;
  size_t off = 0;
  auto alloc = [&](size_t bytes) -> void* {
    void* p = ws + off;
    off += (bytes + 255) & ~(size_t)255;
    return p;
  };

  const size_t KVH = (size_t)NKV * HD;  // 512
  float* Qf = (float*)alloc((size_t)MROWS * HIDDEN * 4);
  float* Kf = (float*)alloc((size_t)MROWS * KVH * 4);
  float* Vf = (float*)alloc((size_t)MROWS * KVH * 4);
  unsigned short* Xb  = (unsigned short*)alloc((size_t)MROWS * HIDDEN * 2);
  unsigned short* Wqt = (unsigned short*)alloc((size_t)HIDDEN * HIDDEN * 2);
  unsigned short* Wkt = (unsigned short*)alloc((size_t)HIDDEN * KVH * 2);
  unsigned short* Wvt = (unsigned short*)alloc((size_t)HIDDEN * KVH * 2);
  unsigned short* Wot = (unsigned short*)alloc((size_t)HIDDEN * HIDDEN * 2);
  unsigned short* Qr  = (unsigned short*)alloc((size_t)MROWS * HIDDEN * 2);
  unsigned short* Kr  = (unsigned short*)alloc((size_t)MROWS * KVH * 2);
  unsigned short* Vtb = (unsigned short*)alloc((size_t)MROWS * KVH * 2);
  unsigned short* Ob  = (unsigned short*)alloc((size_t)MROWS * HIDDEN * 2);

  // 1) operand conversion; weights transposed to [N,K] for TN GEMM
  cvt_bf16  <<<1024, 256, 0, stream>>>(X,  Xb, (size_t)MROWS * HIDDEN);
  cvt_bf16_t<<<1024, 256, 0, stream>>>(Wq, Wqt, HIDDEN, HIDDEN);
  cvt_bf16_t<<<1024, 256, 0, stream>>>(Wk, Wkt, HIDDEN, (int)KVH);
  cvt_bf16_t<<<1024, 256, 0, stream>>>(Wv, Wvt, HIDDEN, (int)KVH);
  cvt_bf16_t<<<1024, 256, 0, stream>>>(Wo, Wot, HIDDEN, HIDDEN);

  // 2) QKV projections (WMMA, async double-buffered tiles)
  gemm_bf16_tn<<<dim3(MROWS / 64, HIDDEN / 128), 256, 0, stream>>>(Xb, Wqt, Qf, MROWS, HIDDEN, HIDDEN);
  gemm_bf16_tn<<<dim3(MROWS / 64, (int)KVH / 128), 256, 0, stream>>>(Xb, Wkt, Kf, MROWS, (int)KVH, HIDDEN);
  gemm_bf16_tn<<<dim3(MROWS / 64, (int)KVH / 128), 256, 0, stream>>>(Xb, Wvt, Vf, MROWS, (int)KVH, HIDDEN);

  // 3) RoPE + bf16; Q scaled by 1/sqrt(128); V written d-major for TDM tiles
  const float qscale = 0.088388347648318447f;
  int totQ = BATCH * SEQ * NH * 64;
  int totK = BATCH * SEQ * NKV * 64;
  rope_cvt<<<(totQ + 255) / 256, 256, 0, stream>>>(Qf, Qr,  NH,  qscale, 1, 0, totQ);
  rope_cvt<<<(totK + 255) / 256, 256, 0, stream>>>(Kf, Kr,  NKV, 1.0f,   1, 0, totK);
  rope_cvt<<<(totK + 255) / 256, 256, 0, stream>>>(Vf, Vtb, NKV, 1.0f,   0, 1, totK);

  // 4) causal flash attention (TDM-staged K/V)
  attn_fwd<<<dim3(SEQ / 64, NH, BATCH), 128, 0, stream>>>(Qr, Kr, Vtb, Ob);

  // 5) output projection
  gemm_bf16_tn<<<dim3(MROWS / 64, HIDDEN / 128), 256, 0, stream>>>(Ob, Wot, out, MROWS, HIDDEN, HIDDEN);
}
